// InterMultiHeadAttention_78847009620033
// MI455X (gfx1250) — matrix operand order repaired and checked by reading, not compiled, not run on hardware
//
#include <hip/hip_runtime.h>
#include <hip/hip_bf16.h>

typedef __attribute__((ext_vector_type(16))) __bf16 v16bf;
typedef __attribute__((ext_vector_type(8)))  float  v8f;

#define K_TOK   2048
#define D_MODEL 1024
#define N_HEADS 16
#define D_HEAD  64
#define FC      32

#define WMMA_BF16(A, B, C) \
  __builtin_amdgcn_wmma_f32_16x16x32_bf16(false, (A), false, (B), (short)0, (C), false, false)

__device__ __forceinline__ unsigned short f2bf(float f) {
  union { float f; unsigned u; } a; a.f = f;
  unsigned u = a.u + 0x7FFFu + ((a.u >> 16) & 1u);   // round-to-nearest-even
  return (unsigned short)(u >> 16);
}

union Frag16 { uint4 q[2]; unsigned short s[16]; v16bf v; };

// ---------------------------------------------------------------- f32 -> bf16
__global__ __launch_bounds__(256) void cvt_bf16_kernel(
    const float* __restrict__ in, unsigned short* __restrict__ out, int n4) {
  int i = blockIdx.x * blockDim.x + threadIdx.x;
  if (i >= n4) return;
  float4 f = ((const float4*)in)[i];
  ushort4 o;
  o.x = f2bf(f.x); o.y = f2bf(f.y); o.z = f2bf(f.z); o.w = f2bf(f.w);
  ((ushort4*)out)[i] = o;
}

// ------------------------------------------- W (KxN f32) -> Wt (NxK bf16)
__global__ __launch_bounds__(256) void transpose_w_kernel(
    const float* __restrict__ W, unsigned short* __restrict__ Wt, int Kd, int N) {
  int k = blockIdx.x;
  for (int n = threadIdx.x; n < N; n += blockDim.x)
    Wt[n * Kd + k] = f2bf(W[k * N + n]);
}

// ------------------------------------------- V (2048x1024 bf16) -> Vt (1024x2048)
__global__ __launch_bounds__(256) void transpose_v_kernel(
    const unsigned short* __restrict__ V, unsigned short* __restrict__ Vt) {
  int r = blockIdx.x;
  for (int c = threadIdx.x; c < D_MODEL; c += blockDim.x)
    Vt[c * K_TOK + r] = V[r * D_MODEL + c];
}

// ---------------------------------------------------------------- bf16 GEMM
// C(MxN) = A(MxKd) * Bt(NxKd)^T + bias.
// One wave per 16x64 output strip: A-frag reused across 4 WMMAs per k-step.
__global__ __launch_bounds__(256) void gemm_bf16_kernel(
    const unsigned short* __restrict__ A,
    const unsigned short* __restrict__ Bt,
    const float* __restrict__ bias,
    void* __restrict__ Cout, int M, int N, int Kd, int out_bf16) {
  const int lane = threadIdx.x & 31;
  const int wid = blockIdx.x * 8 + (threadIdx.x >> 5);
  const int ngrp = N >> 6;                       // 16x64 strips per row
  const int tm = wid / ngrp, tg = wid % ngrp;
  if (tm >= (M >> 4)) return;
  const int l15 = lane & 15, lh = lane >> 4;
  const long arow  = (long)(tm * 16 + l15) * Kd;
  const long brow0 = (long)(tg * 64 + l15) * Kd;
  const long bs = (long)16 * Kd;                 // stride between B n-tiles
  const int ka = 16 * lh;                        // A-frag: contiguous 16 at ka
  const int kb = 16 * lh;                        // B-frag: contiguous 16 at kb
  v8f acc0 = {}, acc1 = {}, acc2 = {}, acc3 = {};
  for (int k0 = 0; k0 < Kd; k0 += 32) {
    // speculative prefetch of next chunk (OOB prefetch is silently dropped)
    __builtin_prefetch((const void*)(A + arow + k0 + 512), 0, 1);
    __builtin_prefetch((const void*)(Bt + brow0 + k0 + 512), 0, 1);
    Frag16 a, b0, b1, b2, b3;
    a.q[0]  = *(const uint4*)(A + arow + k0 + ka);
    a.q[1]  = *(const uint4*)(A + arow + k0 + ka + 8);
    b0.q[0] = *(const uint4*)(Bt + brow0 + k0 + kb);
    b0.q[1] = *(const uint4*)(Bt + brow0 + k0 + kb + 8);
    b1.q[0] = *(const uint4*)(Bt + brow0 + bs + k0 + kb);
    b1.q[1] = *(const uint4*)(Bt + brow0 + bs + k0 + kb + 8);
    b2.q[0] = *(const uint4*)(Bt + brow0 + 2 * bs + k0 + kb);
    b2.q[1] = *(const uint4*)(Bt + brow0 + 2 * bs + k0 + kb + 8);
    b3.q[0] = *(const uint4*)(Bt + brow0 + 3 * bs + k0 + kb);
    b3.q[1] = *(const uint4*)(Bt + brow0 + 3 * bs + k0 + kb + 8);
    acc0 = WMMA_BF16(a.v, b0.v, acc0);
    acc1 = WMMA_BF16(a.v, b1.v, acc1);
    acc2 = WMMA_BF16(a.v, b2.v, acc2);
    acc3 = WMMA_BF16(a.v, b3.v, acc3);
  }
  const int mbase = tm * 16 + 8 * lh;
  v8f accs[4] = {acc0, acc1, acc2, acc3};
#pragma unroll
  for (int t = 0; t < 4; ++t) {
    const int n = tg * 64 + t * 16 + l15;
    const float bv = bias ? bias[n] : 0.0f;
    if (out_bf16) {
      unsigned short* C = (unsigned short*)Cout;
#pragma unroll
      for (int r = 0; r < 8; ++r) C[(long)(mbase + r) * N + n] = f2bf(accs[t][r] + bv);
    } else {
      float* C = (float*)Cout;
#pragma unroll
      for (int r = 0; r < 8; ++r) C[(long)(mbase + r) * N + n] = accs[t][r] + bv;
    }
  }
}

// -------------------------------- colsum[l] = sum_k exp(-|t_k-t_l| * ||c_k-c_l||)
// (weight matrix is symmetric, so column sums == row sums)
__global__ __launch_bounds__(256) void colsum_kernel(
    const float* __restrict__ ts, const float* __restrict__ cc,
    float* __restrict__ colsum) {
  __shared__ float cl[FC];
  __shared__ float rbuf[256];
  const int l = blockIdx.x;
  if (threadIdx.x < FC) cl[threadIdx.x] = cc[l * FC + threadIdx.x];
  __syncthreads();
  const float tl = ts[l];
  float sum = 0.0f;
  for (int k = threadIdx.x; k < K_TOK; k += 256) {
    float dt = fabsf(ts[k] - tl);
    float d2 = 0.0f;
#pragma unroll
    for (int f = 0; f < FC; ++f) { float df = cc[k * FC + f] - cl[f]; d2 += df * df; }
    sum += __expf(-dt * sqrtf(d2));
  }
  rbuf[threadIdx.x] = sum;
  __syncthreads();
  for (int s = 128; s > 0; s >>= 1) {
    if (threadIdx.x < s) rbuf[threadIdx.x] += rbuf[threadIdx.x + s];
    __syncthreads();
  }
  if (threadIdx.x == 0) colsum[l] = rbuf[0];
}

// -------------------------------- nw[k][l] = exp(-dt*dc)/colsum[l]  (16MB, L2-resident)
__global__ __launch_bounds__(256) void normw_kernel(
    const float* __restrict__ ts, const float* __restrict__ cc,
    const float* __restrict__ colsum, float* __restrict__ nw) {
  __shared__ float ck[FC];
  const int k = blockIdx.x;
  if (threadIdx.x < FC) ck[threadIdx.x] = cc[k * FC + threadIdx.x];
  __syncthreads();
  const float tk = ts[k];
  for (int l = threadIdx.x; l < K_TOK; l += 256) {
    float dt = fabsf(ts[l] - tk);
    float d2 = 0.0f;
#pragma unroll
    for (int f = 0; f < FC; ++f) { float df = cc[l * FC + f] - ck[f]; d2 += df * df; }
    nw[(long)k * K_TOK + l] = __expf(-dt * sqrtf(d2)) / colsum[l];
  }
}

// ---------------------------------------------------------------- fused attention
// One workgroup = (head h, 16-row query block). 8 waves.
// Stage 1: S[16][2048] = (Q Kt / sqrt(2048)) * nw   (WMMA, C-frag spill to LDS)
// Stage 2: row softmax over LDS
// Stage 3: O = P V via WMMA (P re-packed f32->bf16 from LDS, V from Vt), ds_add_f32 merge
__global__ __launch_bounds__(256) void attn_kernel(
    const unsigned short* __restrict__ Qp,
    const unsigned short* __restrict__ Kp,
    const unsigned short* __restrict__ Vt,
    const float* __restrict__ nw,
    unsigned short* __restrict__ attn) {
  extern __shared__ char smem[];
  float* S    = (float*)smem;                                  // 16*2048 f32
  float* red  = (float*)(smem + 16 * K_TOK * sizeof(float));   // 16*16
  float* Oacc = red + 16 * 16;                                 // 16*64

  const int tid = threadIdx.x;
  const int lane = tid & 31, w = tid >> 5;
  const int h = blockIdx.x >> 7, mb = blockIdx.x & 127;
  const int m0 = mb * 16;
  const int l15 = lane & 15, lh = lane >> 4;
  const int ka = 16 * lh, kb = 16 * lh;

  // ---- Stage 1: scores
  Frag16 a0, a1;
  {
    const unsigned short* qrow = Qp + (long)(m0 + l15) * D_MODEL + h * D_HEAD;
    a0.q[0] = *(const uint4*)(qrow + ka);
    a0.q[1] = *(const uint4*)(qrow + ka + 8);
    a1.q[0] = *(const uint4*)(qrow + 32 + ka);
    a1.q[1] = *(const uint4*)(qrow + 32 + ka + 8);
  }
  const float invs = 0.02209708691207961f;  // 1/sqrt(2048)
  for (int i = 0; i < 16; ++i) {
    const int nt = w * 16 + i * 128;
    const unsigned short* krow = Kp + (long)(nt + l15) * D_MODEL + h * D_HEAD;
    Frag16 b0, b1;
    b0.q[0] = *(const uint4*)(krow + kb);
    b0.q[1] = *(const uint4*)(krow + kb + 8);
    b1.q[0] = *(const uint4*)(krow + 32 + kb);
    b1.q[1] = *(const uint4*)(krow + 32 + kb + 8);
    v8f acc = {};
    acc = WMMA_BF16(a0.v, b0.v, acc);
    acc = WMMA_BF16(a1.v, b1.v, acc);
    const int n = nt + l15;
#pragma unroll
    for (int r = 0; r < 8; ++r) {
      const int m = 8 * lh + r;                 // C layout: M = r + 8*(lane>>4)
      S[m * K_TOK + n] = acc[r] * invs * nw[(long)(m0 + m) * K_TOK + n];
    }
  }
  __syncthreads();

  // ---- Stage 2: softmax (16 threads per row)
  {
    const int row = tid >> 4, c0 = tid & 15;
    float* Srow = S + row * K_TOK;
    float mx = -3.4028235e38f;
    for (int c = c0; c < K_TOK; c += 16) mx = fmaxf(mx, Srow[c]);
    red[row * 16 + c0] = mx;
    __syncthreads();
    mx = red[row * 16];
    for (int j = 1; j < 16; ++j) mx = fmaxf(mx, red[row * 16 + j]);
    float sum = 0.0f;
    for (int c = c0; c < K_TOK; c += 16) {
      float e = __expf(Srow[c] - mx);
      Srow[c] = e;
      sum += e;
    }
    __syncthreads();
    red[row * 16 + c0] = sum;
    __syncthreads();
    sum = 0.0f;
    for (int j = 0; j < 16; ++j) sum += red[row * 16 + j];
    const float inv = 1.0f / sum;
    for (int c = c0; c < K_TOK; c += 16) Srow[c] *= inv;
  }
  for (int idx = tid; idx < 16 * 64; idx += 256) Oacc[idx] = 0.0f;
  __syncthreads();

  // ---- Stage 3: O = P @ V
  {
    const int ct = w & 3, half = w >> 2;   // wave -> (col tile, K half)
    const unsigned short* vrow = Vt + (long)(h * D_HEAD + ct * 16 + l15) * K_TOK;
    v8f o = {};
    for (int lc = half * 1024; lc < half * 1024 + 1024; lc += 32) {
      __builtin_prefetch((const void*)(vrow + lc + 256), 0, 1);
      Frag16 a, b;
      const float* sp = S + l15 * K_TOK + lc + ka;
#pragma unroll
      for (int j = 0; j < 8; ++j) a.s[j] = f2bf(sp[j]);
#pragma unroll
      for (int j = 0; j < 8; ++j) a.s[8 + j] = f2bf(sp[8 + j]);
      b.q[0] = *(const uint4*)(vrow + lc + kb);
      b.q[1] = *(const uint4*)(vrow + lc + kb + 8);
      o = WMMA_BF16(a.v, b.v, o);
    }
    const int n = ct * 16 + l15;
#pragma unroll
    for (int r = 0; r < 8; ++r)
      atomicAdd(&Oacc[(8 * lh + r) * 64 + n], o[r]);   // ds_add_f32 merge of K-halves
  }
  __syncthreads();
  for (int idx = tid; idx < 16 * 64; idx += 256) {
    const int m = idx >> 6, v = idx & 63;
    attn[(long)(m0 + m) * D_MODEL + h * D_HEAD + v] = f2bf(Oacc[idx]);
  }
}

// ================================================================ host side
extern "C" void kernel_launch(void* const* d_in, const int* in_sizes, int n_in,
                              void* d_out, int out_size, void* d_ws, size_t ws_size,
                              hipStream_t stream) {
  (void)in_sizes; (void)n_in; (void)out_size; (void)ws_size;
  const float* q  = (const float*)d_in[0];
  const float* k  = (const float*)d_in[1];
  const float* v  = (const float*)d_in[2];
  const float* ts = (const float*)d_in[3];
  const float* cc = (const float*)d_in[4];
  const float* Wq = (const float*)d_in[5];
  const float* bq = (const float*)d_in[6];
  const float* Wk = (const float*)d_in[7];
  const float* bk = (const float*)d_in[8];
  const float* Wv = (const float*)d_in[9];
  const float* bv = (const float*)d_in[10];
  const float* Wo = (const float*)d_in[11];
  const float* bo = (const float*)d_in[12];

  char* ws = (char*)d_ws;
  size_t off = 0;
  auto carve = [&](size_t bytes) -> char* {
    char* p = ws + off;
    off += (bytes + 255) & ~(size_t)255;
    return p;
  };
  const size_t XB = (size_t)K_TOK * D_MODEL * 2;   // 4 MB bf16
  const size_t WB = (size_t)D_MODEL * D_MODEL * 2; // 2 MB bf16
  unsigned short* Xq  = (unsigned short*)carve(XB);
  unsigned short* Xk  = (unsigned short*)carve(XB);
  unsigned short* Xv  = (unsigned short*)carve(XB);
  unsigned short* Wqt = (unsigned short*)carve(WB);
  unsigned short* Wkt = (unsigned short*)carve(WB);
  unsigned short* Wvt = (unsigned short*)carve(WB);
  unsigned short* Wot = (unsigned short*)carve(WB);
  unsigned short* Qp  = (unsigned short*)carve(XB);
  unsigned short* Kp  = (unsigned short*)carve(XB);
  unsigned short* Vp  = (unsigned short*)carve(XB);
  unsigned short* Vtr = (unsigned short*)carve(XB);
  unsigned short* Ab  = (unsigned short*)carve(XB);                       // attention out bf16
  float* cs = (float*)carve((size_t)K_TOK * sizeof(float));               // colsum
  float* nw = (float*)carve((size_t)K_TOK * K_TOK * sizeof(float));       // 16 MB

  const int n4 = (K_TOK * D_MODEL) / 4;
  cvt_bf16_kernel<<<n4 / 256, 256, 0, stream>>>(q, Xq, n4);
  cvt_bf16_kernel<<<n4 / 256, 256, 0, stream>>>(k, Xk, n4);
  cvt_bf16_kernel<<<n4 / 256, 256, 0, stream>>>(v, Xv, n4);

  transpose_w_kernel<<<D_MODEL, 256, 0, stream>>>(Wq, Wqt, D_MODEL, D_MODEL);
  transpose_w_kernel<<<D_MODEL, 256, 0, stream>>>(Wk, Wkt, D_MODEL, D_MODEL);
  transpose_w_kernel<<<D_MODEL, 256, 0, stream>>>(Wv, Wvt, D_MODEL, D_MODEL);
  transpose_w_kernel<<<D_MODEL, 256, 0, stream>>>(Wo, Wot, D_MODEL, D_MODEL);

  const int gemm_blocks = (K_TOK / 16) * (D_MODEL / 64) / 8;  // 256 blocks, 8 waves each
  gemm_bf16_kernel<<<gemm_blocks, 256, 0, stream>>>(Xq, Wqt, bq, Qp, K_TOK, D_MODEL, D_MODEL, 1);
  gemm_bf16_kernel<<<gemm_blocks, 256, 0, stream>>>(Xk, Wkt, bk, Kp, K_TOK, D_MODEL, D_MODEL, 1);
  gemm_bf16_kernel<<<gemm_blocks, 256, 0, stream>>>(Xv, Wvt, bv, Vp, K_TOK, D_MODEL, D_MODEL, 1);

  transpose_v_kernel<<<K_TOK, 256, 0, stream>>>(Vp, Vtr);

  colsum_kernel<<<K_TOK, 256, 0, stream>>>(ts, cc, cs);
  normw_kernel<<<K_TOK, 256, 0, stream>>>(ts, cc, cs, nw);

  const size_t shmem = 16 * K_TOK * sizeof(float) + 16 * 16 * sizeof(float) +
                       16 * 64 * sizeof(float);  // ~133 KB (< 320 KB/WGP)
  attn_kernel<<<N_HEADS * (K_TOK / 16), 256, shmem, stream>>>(Qp, Kp, Vtr, nw, Ab);

  gemm_bf16_kernel<<<gemm_blocks, 256, 0, stream>>>(Ab, Wot, bo, (float*)d_out,
                                                    K_TOK, D_MODEL, D_MODEL, 0);
}
